// MultiHeadAttention_43404939493787
// MI455X (gfx1250) — matrix operand order repaired and checked by reading, not compiled, not run on hardware
//
#include <hip/hip_runtime.h>
#include <hip/hip_bf16.h>
#include <math.h>

// ---------------------------------------------------------------------------
// MHA for MI455X (gfx1250): bf16 WMMA GEMMs + flash-attention streaming core.
// Round 3: row-max reduction via DPP (quad_perm/row_mirror) instead of
// shfl->ds_bpermute; row-sum via WMMA-with-ones; KB=64; GEMM BK=64.
// ---------------------------------------------------------------------------

typedef __bf16 bf16_t;
typedef __attribute__((ext_vector_type(16))) __bf16 v16bf;
typedef __attribute__((ext_vector_type(8)))  __bf16 v8bf;
typedef __attribute__((ext_vector_type(4)))  __bf16 v4bf;
typedef __attribute__((ext_vector_type(8)))  float  v8f;
typedef __attribute__((ext_vector_type(4)))  float  v4f;

union BF16x16 { v16bf v; v8bf h[2]; };
union FI32 { float f; int i; };

// Max over the 16-lane DPP row, pure VALU (no LDS/bpermute).
// Masks generated: quad_perm^1, quad_perm^2, row_half_mirror(^7),
// row_mirror(^15) -> span covers all 16 lanes; max is idempotent.
static __device__ inline float dpp_rowmax16(float x) {
  FI32 a, t; a.f = x;
  t.i = __builtin_amdgcn_update_dpp(0, a.i, 0xB1,  0xf, 0xf, true); // [1,0,3,2]
  a.f = fmaxf(a.f, t.f);
  t.i = __builtin_amdgcn_update_dpp(0, a.i, 0x4E,  0xf, 0xf, true); // [2,3,0,1]
  a.f = fmaxf(a.f, t.f);
  t.i = __builtin_amdgcn_update_dpp(0, a.i, 0x141, 0xf, 0xf, true); // half_mirror
  a.f = fmaxf(a.f, t.f);
  t.i = __builtin_amdgcn_update_dpp(0, a.i, 0x140, 0xf, 0xf, true); // row_mirror
  a.f = fmaxf(a.f, t.f);
  return a.f;
}

// A-fragment (16x32 bf16, row-major source, stride lda elements):
// lanes 0-15: row=lane,    K chunks {0..7, 16..23}
// lanes16-31: row=lane-16, K chunks {8..15, 24..31}
static __device__ inline v16bf load_a_frag_lds(const bf16_t* smem, int lda) {
  const int lane = threadIdx.x & 31;
  const int row  = lane & 15;
  const int kb   = (lane >> 4) << 3;   // 0 or 8
  const bf16_t* p = smem + row * lda + kb;
  BF16x16 u;
  u.h[0] = *(const v8bf*)(p);
  u.h[1] = *(const v8bf*)(p + 16);
  return u.v;
}

// B-fragment (32x16 bf16), source with rows indexed by the N (column) index
// and K contiguous: lanes 0-15 hold K chunks {0..7, 16..23} of column lane;
// lanes 16-31 hold K chunks {8..15, 24..31} of column lane-16.
static __device__ inline v16bf load_b_frag(const bf16_t* base, int row_stride) {
  const int lane = threadIdx.x & 31;
  const bf16_t* p = base + (size_t)(lane & 15) * row_stride + ((lane >> 4) << 3);
  BF16x16 u;
  u.h[0] = *(const v8bf*)(p);
  u.h[1] = *(const v8bf*)(p + 16);
  return u.v;
}

static __device__ inline v8f wmma_bf16(v16bf a, v16bf b, v8f c) {
  return __builtin_amdgcn_wmma_f32_16x16x32_bf16(
      /*neg_a=*/false, a, /*neg_b=*/false, b,
      /*c_mod=*/(short)0, c, /*reuse_a=*/false, /*reuse_b=*/false);
}

// ---------------------------------------------------------------------------
// Y = A(8192 x 512) @ W(512 x 512) + bias, scaled by out_scale.
// OUT_MODE 0: fp32 row-major (final projection)
// OUT_MODE 1: bf16 head-split       dst[((b*8+h)*4096+s)*64 + d]
// OUT_MODE 2: bf16 head-split V^T   dst[((b*8+h)*64+d)*4096 + s]
// Block: 256 threads (8 waves); tile 128x64; wave tile 32x32; BK=64.
// ---------------------------------------------------------------------------
template <typename AT, int OUT_MODE>
__global__ __launch_bounds__(256)
void gemm512_wmma(const AT* __restrict__ A, const float* __restrict__ W,
                  const float* __restrict__ bias, void* __restrict__ Yv,
                  float out_scale)
{
  __shared__ bf16_t As[128 * 72];   // 128 rows x 64 K (+8 pad), bf16
  __shared__ bf16_t Bs[64 * 72];    // 64 cols(n) x 64 K (transposed W tile)

  const int rowBase = blockIdx.x * 128;
  const int colBase = blockIdx.y * 64;
  const int tid  = threadIdx.x;
  const int wave = tid >> 5;
  const int wm = wave >> 1;         // 0..3
  const int wn = wave & 1;          // 0..1

  v8f acc[2][2] = {};

  for (int k0 = 0; k0 < 512; k0 += 64) {
    // ---- stage A tile (128x64), converting to bf16 ----
    #pragma unroll
    for (int p = 0; p < 8; ++p) {
      int idx = p * 1024 + tid * 4;
      int r  = idx >> 6;
      int kk = idx & 63;
      const AT* src = A + (size_t)(rowBase + r) * 512 + k0 + kk;
      bf16_t* dst = &As[r * 72 + kk];
      if constexpr (sizeof(AT) == 4) {
        v4f x = *(const v4f*)src;
        dst[0] = (bf16_t)x.x; dst[1] = (bf16_t)x.y;
        dst[2] = (bf16_t)x.z; dst[3] = (bf16_t)x.w;
      } else {
        *(v4bf*)dst = *(const v4bf*)src;
      }
    }
    // ---- stage W tile transposed: Bs[n][kk] = W[k0+kk][colBase+n] ----
    #pragma unroll
    for (int p = 0; p < 4; ++p) {
      int idx = p * 1024 + tid * 4;
      int kk = idx >> 6;
      int n  = idx & 63;
      v4f wv = *(const v4f*)(W + (size_t)(k0 + kk) * 512 + colBase + n);
      Bs[(n + 0) * 72 + kk] = (bf16_t)wv.x;
      Bs[(n + 1) * 72 + kk] = (bf16_t)wv.y;
      Bs[(n + 2) * 72 + kk] = (bf16_t)wv.z;
      Bs[(n + 3) * 72 + kk] = (bf16_t)wv.w;
    }
    __syncthreads();

    #pragma unroll
    for (int kf = 0; kf < 2; ++kf) {
      v16bf a0 = load_a_frag_lds(&As[(wm * 32)      * 72 + kf * 32], 72);
      v16bf a1 = load_a_frag_lds(&As[(wm * 32 + 16) * 72 + kf * 32], 72);
      v16bf b0 = load_b_frag(&Bs[(wn * 32)      * 72] + kf * 32, 72);
      v16bf b1 = load_b_frag(&Bs[(wn * 32 + 16) * 72] + kf * 32, 72);
      acc[0][0] = wmma_bf16(a0, b0, acc[0][0]);
      acc[0][1] = wmma_bf16(a0, b1, acc[0][1]);
      acc[1][0] = wmma_bf16(a1, b0, acc[1][0]);
      acc[1][1] = wmma_bf16(a1, b1, acc[1][1]);
    }
    __syncthreads();
  }

  // ---- epilogue: VGPR r holds row r+8*(lane>=16), col lane&15 ----
  const int lane = tid & 31;
  const int half = lane >> 4, nl = lane & 15;
  #pragma unroll
  for (int ti = 0; ti < 2; ++ti)
    #pragma unroll
    for (int tj = 0; tj < 2; ++tj)
      #pragma unroll
      for (int r = 0; r < 8; ++r) {
        int gm = rowBase + wm * 32 + ti * 16 + half * 8 + r;
        int gn = colBase + wn * 32 + tj * 16 + nl;
        float v = (acc[ti][tj][r] + bias[gn]) * out_scale;
        if constexpr (OUT_MODE == 0) {
          ((float*)Yv)[(size_t)gm * 512 + gn] = v;
        } else {
          int bb = gm >> 12, s = gm & 4095;
          int h  = gn >> 6,  d = gn & 63;
          bf16_t* Y = (bf16_t*)Yv;
          if constexpr (OUT_MODE == 1)
            Y[(((size_t)(bb * 8 + h)) * 4096 + s) * 64 + d] = (bf16_t)v;
          else
            Y[(((size_t)(bb * 8 + h)) * 64 + d) * 4096 + s] = (bf16_t)v;
        }
      }
}

// ---------------------------------------------------------------------------
// Flash attention: grid (S/128, B*H); 8 waves/block; one 16-row Q strip/wave.
// KB=64 keys per iteration. Row-max via DPP (VALU only); row-sum via WMMA
// against an all-ones B matrix (replicated across lanes in C-layout).
// ---------------------------------------------------------------------------
__global__ __launch_bounds__(256)
void flash_attn_wmma(const bf16_t* __restrict__ Qh, const bf16_t* __restrict__ Kh,
                     const bf16_t* __restrict__ Vt, const float* __restrict__ mask,
                     bf16_t* __restrict__ ctx)
{
  constexpr int S = 4096, D = 64;
  __shared__ bf16_t Ps[8 * 16 * 72];   // per-wave 16x64 P scratch (+pad)

  const int bh = blockIdx.y;            // b*8 + h
  const int b  = bh >> 3;
  const int h  = bh & 7;
  const int wave = threadIdx.x >> 5;
  const int lane = threadIdx.x & 31;
  const int half = lane >> 4, nl = lane & 15;
  const int qbase = blockIdx.x * 128 + wave * 16;

  const bf16_t* Qp = Qh + (size_t)bh * S * D;
  const bf16_t* Kp = Kh + (size_t)bh * S * D;
  const bf16_t* Vp = Vt + (size_t)bh * D * S;   // (depth, S) transposed
  const float*  mp = mask + (size_t)b * S;
  bf16_t* ps = &Ps[wave * 16 * 72];

  // All-ones B fragment for row-sum WMMA.
  v16bf onesf;
  #pragma unroll
  for (int i = 0; i < 16; ++i) onesf[i] = (bf16_t)1.0f;

  // Q A-fragments: 16 rows x 64 depth -> two 16x32 frags, direct from global.
  v16bf qf[2];
  {
    const bf16_t* base = Qp + (size_t)(qbase + nl) * D + half * 8;
    #pragma unroll
    for (int f = 0; f < 2; ++f) {
      BF16x16 u;
      u.h[0] = *(const v8bf*)(base + f * 32);
      u.h[1] = *(const v8bf*)(base + f * 32 + 16);
      qf[f] = u.v;
    }
  }

  v8f o[4] = {};
  float mrow[8], lrow[8];
  #pragma unroll
  for (int r = 0; r < 8; ++r) { mrow[r] = -3.0e38f; lrow[r] = 0.0f; }

  for (int kb = 0; kb < S; kb += 64) {
    // ---- S tile = Q . K^T  (16 x 64) : four 16x16 C tiles, 8 WMMAs ----
    v8f st[4];
    #pragma unroll
    for (int t = 0; t < 4; ++t) {
      v8f z = {};
      v16bf kf0 = load_b_frag(Kp + (size_t)(kb + t * 16) * D,      D);
      v16bf kf1 = load_b_frag(Kp + (size_t)(kb + t * 16) * D + 32, D);
      z = wmma_bf16(qf[0], kf0, z);
      z = wmma_bf16(qf[1], kf1, z);
      float mv = mp[kb + t * 16 + nl] * -1.0e9f;
      #pragma unroll
      for (int r = 0; r < 8; ++r) z[r] += mv;
      st[t] = z;
    }

    // ---- row max: VALU over 4 tiles, then DPP row reduction (no LDS) ----
    float rmax[8];
    #pragma unroll
    for (int r = 0; r < 8; ++r) {
      rmax[r] = fmaxf(fmaxf(st[0][r], st[1][r]), fmaxf(st[2][r], st[3][r]));
      rmax[r] = dpp_rowmax16(rmax[r]);
    }

    // ---- online softmax update; write P (bf16) into per-wave LDS ----
    float sc[8];
    #pragma unroll
    for (int r = 0; r < 8; ++r) {
      float nm = fmaxf(mrow[r], rmax[r]);
      sc[r] = __expf(mrow[r] - nm);
      mrow[r] = nm;
      lrow[r] *= sc[r];
      o[0][r] *= sc[r]; o[1][r] *= sc[r]; o[2][r] *= sc[r]; o[3][r] *= sc[r];
      int prow = (half * 8 + r) * 72;
      #pragma unroll
      for (int t = 0; t < 4; ++t) {
        float pv = __expf(st[t][r] - nm);
        ps[prow + t * 16 + nl] = (bf16_t)pv;
      }
    }
    asm volatile("s_wait_dscnt 0" ::: "memory");   // wave-local cross-lane LDS
    v16bf pf0 = load_a_frag_lds(ps,      72);      // keys kb..kb+31
    v16bf pf1 = load_a_frag_lds(ps + 32, 72);      // keys kb+32..kb+63

    // ---- row sum via WMMA with all-ones B (replicated across lanes) ----
    v8f lacc = {};
    lacc = wmma_bf16(pf0, onesf, lacc);
    lacc = wmma_bf16(pf1, onesf, lacc);
    #pragma unroll
    for (int r = 0; r < 8; ++r) lrow[r] += lacc[r];

    // ---- O += P . V  (V^T layout: lane=depth col, keys contiguous) ----
    #pragma unroll
    for (int t = 0; t < 4; ++t) {
      v16bf vf0 = load_b_frag(Vp + (size_t)(t * 16) * S + kb,      S);
      v16bf vf1 = load_b_frag(Vp + (size_t)(t * 16) * S + kb + 32, S);
      o[t] = wmma_bf16(pf0, vf0, o[t]);
      o[t] = wmma_bf16(pf1, vf1, o[t]);
    }
  }

  // ---- normalize and write context, concat-head layout (B, S, 512), bf16 ----
  #pragma unroll
  for (int t = 0; t < 4; ++t)
    #pragma unroll
    for (int r = 0; r < 8; ++r) {
      int s = qbase + half * 8 + r;
      int d = t * 16 + nl;
      ctx[((size_t)(b * 4096 + s)) * 512 + h * 64 + d] = (bf16_t)(o[t][r] / lrow[r]);
    }
}

// ---------------------------------------------------------------------------
extern "C" void kernel_launch(void* const* d_in, const int* in_sizes, int n_in,
                              void* d_out, int out_size, void* d_ws, size_t ws_size,
                              hipStream_t stream)
{
  const float* query = (const float*)d_in[0];
  const float* key   = (const float*)d_in[1];
  const float* value = (const float*)d_in[2];
  const float* mask  = (const float*)d_in[3];
  const float* Wq = (const float*)d_in[4];
  const float* bq = (const float*)d_in[5];
  const float* Wk = (const float*)d_in[6];
  const float* bk = (const float*)d_in[7];
  const float* Wv = (const float*)d_in[8];
  const float* bv = (const float*)d_in[9];
  const float* Wo = (const float*)d_in[10];
  const float* bo = (const float*)d_in[11];

  const size_t HE = (size_t)2 * 8 * 4096 * 64;  // per-tensor head-split elems
  bf16_t* Qh  = (bf16_t*)d_ws;       // (B,H,S,64)  bf16, pre-scaled by 1/8
  bf16_t* Kh  = Qh + HE;             // (B,H,S,64)  bf16
  bf16_t* Vt  = Kh + HE;             // (B,H,64,S)  bf16 (transposed)
  bf16_t* ctx = Vt + HE;             // (B,S,512)   bf16
  (void)ws_size; (void)in_sizes; (void)n_in; (void)out_size;

  dim3 blk(256);
  dim3 gG(64, 8);                    // 8192/128 x 512/64

  gemm512_wmma<float, 1><<<gG, blk, 0, stream>>>(query, Wq, bq, Qh, 0.125f);
  gemm512_wmma<float, 1><<<gG, blk, 0, stream>>>(key,   Wk, bk, Kh, 1.0f);
  gemm512_wmma<float, 2><<<gG, blk, 0, stream>>>(value, Wv, bv, Vt, 1.0f);

  dim3 gA(32, 16);                   // 4096/128 q-tiles x (B*H)
  flash_attn_wmma<<<gA, blk, 0, stream>>>(Qh, Kh, Vt, mask, ctx);

  gemm512_wmma<bf16_t, 0><<<gG, blk, 0, stream>>>(ctx, Wo, bo, d_out, 1.0f);
}
